// RNN_SingleOutput_68908455297449
// MI455X (gfx1250) — matrix operand order repaired and checked by reading, not compiled, not run on hardware
//
#include <hip/hip_runtime.h>
#include <hip/hip_bf16.h>
#include <stdint.h>

// ---------------------------------------------------------------------------
// 2-layer Elman ReLU RNN (B=256,T=512,I=64,H=256) + FC head on gfx1250.
// Latency-bound sequential scan:
//  - weight B-fragments permanently in VGPRs (built once from global f32)
//  - v_wmma_f32_16x16x32_f16, two independent accumulate chains (proj/rec)
//  - A-fragments batch-preloaded, pinned with __builtin_amdgcn_sched_barrier
//    so the machine scheduler cannot re-serialize ds_load -> wmma pairs;
//    the WMMA chain then issues back-to-back with staggered dscnt waits
//  - L0: x_{t+1} register double-buffer issued AFTER the barrier (wait lands
//    next iteration); h1 streamed out with global_store_async_from_lds_b128
//  - L1: h1_t staged one step ahead with global_load_async_to_lds_b128 into
//    ping-pong LDS buffers (s_wait_asynccnt 0x1 => one transfer in flight);
//    FC head fused via butterfly shuffles + ds_add_f32
// Workspace: h1 intermediate f16 [B,T,H] = 64 MiB.
// ---------------------------------------------------------------------------

#define Bsz   256
#define Tlen  512
#define Isz   64
#define Hsz   256
#define HPAD  264   // 256 + 8 f16 (16B) row padding
#define IPAD  72    // 64 + 8 f16 padding
#define NTHREADS 512

typedef __attribute__((ext_vector_type(16))) _Float16 v16h;
typedef __attribute__((ext_vector_type(8)))  float    v8f;

#define WMMA_F16(a, b, c) \
  __builtin_amdgcn_wmma_f32_16x16x32_f16(false, (a), false, (b), (short)0, (c), false, false)

// A fragment (16x32 f16): lane L<16: M=L, K=kc+{0..7,16..23};
// lane L>=16: M=L-16, K=kc+{8..15,24..31}.
static __device__ __forceinline__ v16h load_frag_a(const _Float16* base, int stride,
                                                   int m, int kc, int lane) {
  const _Float16* p = base + m * stride + kc + ((lane < 16) ? 0 : 8);
  union { v16h v; float4 q[2]; } u;
  u.q[0] = *(const float4*)p;
  u.q[1] = *(const float4*)(p + 16);
  return u.v;
}

// B fragment (32x16 f16) built directly from global f32 row-major W[n][k]:
// lane L holds column N = n, K = kc+{0..7,16..23} (L<16) or kc+{8..15,24..31} (L>=16).
// (h @ W^T  =>  B[k][n] = W[n][k], contiguous along k.)
static __device__ __forceinline__ v16h make_frag_b(const float* __restrict__ W, int ldw,
                                                   int n, int kc, int lane) {
  const float* p = W + (size_t)n * ldw + kc + ((lane < 16) ? 0 : 8);
  union { v16h v; _Float16 h[16]; } u;
#pragma unroll
  for (int i = 0; i < 16; ++i) u.h[i] = (_Float16)p[(i < 8) ? i : i + 8];
  return u.v;
}

// ---------------------------------------------------------------------------
// Layer 0: h1_t = relu(W_ih0 x_t + b_ih0 + b_hh0 + W_hh0 h_{t-1})
// 16 WGs (16-batch tiles) x 512 threads (16 waves; wave = one 16-wide N tile).
// ---------------------------------------------------------------------------
__global__ __launch_bounds__(NTHREADS)
void rnn_layer0_kernel(const float* __restrict__ x,
                       const float* __restrict__ Wih,
                       const float* __restrict__ Whh,
                       const float* __restrict__ bih,
                       const float* __restrict__ bhh,
                       _Float16* __restrict__ h1out) {
  __shared__ _Float16 sX[16 * IPAD];        // staged x_t (f16)
  __shared__ _Float16 sH[2][16 * HPAD];     // ping-pong hidden state

  const int tid   = threadIdx.x;
  const int lane  = tid & 31;
  const int bBase = blockIdx.x * 16;
  const int nl    = (tid >> 5) * 16 + (lane & 15);  // hidden column
  const int mrow  = lane & 15;
  const int mbase = (lane < 16) ? 0 : 8;

  for (int i = tid; i < 16 * HPAD; i += NTHREADS) sH[0][i] = (_Float16)0.f;

  // Loop-invariant weight fragments -> VGPRs (80 regs).
  v16h wIh[2], wHh[8];
#pragma unroll
  for (int k = 0; k < 2; ++k) wIh[k] = make_frag_b(Wih, Isz, nl, k * 32, lane);
#pragma unroll
  for (int k = 0; k < 8; ++k) wHh[k] = make_frag_b(Whh, Hsz, nl, k * 32, lane);

  const float biasv = bih[nl] + bhh[nl];

  // x staging: row = wave, 2 f32 per thread; register double-buffer.
  const int xr = tid >> 5, xc = (tid * 2) & 63;
  const float* xrow = x + (size_t)(bBase + xr) * Tlen * Isz + xc;
  float2 xreg = *(const float2*)xrow;       // t = 0

  // write-out lane mapping: row = wave, 8 f16 per lane
  const int cr = tid >> 5;
  const int c8 = lane * 8;

  int p = 0;
  for (int t = 0; t < Tlen; ++t) {
    // Commit staged x_t.
    sX[xr * IPAD + xc]     = (_Float16)xreg.x;
    sX[xr * IPAD + xc + 1] = (_Float16)xreg.y;
    // Async store issued 2 steps ago must be done before sH[p^1] is rewritten;
    // allow the most recent one to stay in flight.
    asm volatile("s_wait_asynccnt 0x1" ::: "memory");
    __syncthreads();

    // Issue x_{t+1} load NOW (after the barrier) so its s_wait_loadcnt lands
    // at the next iteration's commit, hidden under the WMMA chain.
    if (t + 1 < Tlen) {
      xreg = *(const float2*)(xrow + (size_t)(t + 1) * Isz);
      if (t + 2 < Tlen) __builtin_prefetch(xrow + (size_t)(t + 2) * Isz, 0, 1);
    }

    // Batch-preload ALL A-fragments, then pin the schedule: nothing may cross
    // the sched_barrier, so all 20 ds_load_b128 stay in flight together and
    // regalloc must keep the 10 fragments in distinct registers.
    v16h ax0 = load_frag_a(sX, IPAD, mrow, 0, lane);
    v16h ax1 = load_frag_a(sX, IPAD, mrow, 32, lane);
    v16h ar[8];
#pragma unroll
    for (int k = 0; k < 8; ++k)
      ar[k] = load_frag_a(sH[p], HPAD, mrow, k * 32, lane);
    __builtin_amdgcn_sched_barrier(0);

    v8f c0, c1;
#pragma unroll
    for (int r = 0; r < 8; ++r) { c0[r] = biasv; c1[r] = 0.f; }

    c0 = WMMA_F16(ax0, wIh[0], c0);
    c0 = WMMA_F16(ax1, wIh[1], c0);
#pragma unroll
    for (int k = 0; k < 8; ++k)
      c1 = WMMA_F16(ar[k], wHh[k], c1);

    const int q = p ^ 1;
#pragma unroll
    for (int r = 0; r < 8; ++r) {
      float v = fmaxf(c0[r] + c1[r], 0.f);                 // ReLU
      sH[q][(mbase + r) * HPAD + nl] = (_Float16)v;
    }
    __syncthreads();

    // Stream h1_t to HBM straight from LDS (one async instr per wave).
    {
      uint64_t ga = (uint64_t)(uintptr_t)(h1out +
                    ((size_t)(bBase + cr) * Tlen + t) * Hsz + c8);
      uint32_t la = (uint32_t)(uintptr_t)&sH[q][cr * HPAD + c8];
      asm volatile("global_store_async_from_lds_b128 %0, %1, off"
                   :: "v"(ga), "v"(la) : "memory");
    }
    p = q;
  }
  asm volatile("s_wait_asynccnt 0x0" ::: "memory");
}

// ---------------------------------------------------------------------------
// Layer 1 + FC head:
// h2_t = relu(W_ih1 h1_t + b_ih1 + b_hh1 + W_hh1 h2_{t-1});
// out[b*T+t] = dot(h2_t, w_fc) + b_fc.
// ---------------------------------------------------------------------------
__global__ __launch_bounds__(NTHREADS)
void rnn_layer1_kernel(const _Float16* __restrict__ h1,
                       const float* __restrict__ Wih,
                       const float* __restrict__ Whh,
                       const float* __restrict__ bih,
                       const float* __restrict__ bhh,
                       const float* __restrict__ Wfc,
                       const float* __restrict__ bfc,
                       float* __restrict__ out) {
  __shared__ _Float16 sIn[2][16 * HPAD];    // ping-pong staged h1_t
  __shared__ _Float16 sH[2][16 * HPAD];     // ping-pong hidden state
  __shared__ float    sAcc[16];

  const int tid   = threadIdx.x;
  const int lane  = tid & 31;
  const int wave  = tid >> 5;
  const int bBase = blockIdx.x * 16;
  const int nl    = wave * 16 + (lane & 15);
  const int mrow  = lane & 15;
  const int mbase = (lane < 16) ? 0 : 8;

  for (int i = tid; i < 16 * HPAD; i += NTHREADS) sH[0][i] = (_Float16)0.f;
  if (tid < 16) sAcc[tid] = 0.f;

  // Loop-invariant weight fragments -> VGPRs (128 regs).
  v16h wIh[8], wHh[8];
#pragma unroll
  for (int k = 0; k < 8; ++k) wIh[k] = make_frag_b(Wih, Hsz, nl, k * 32, lane);
#pragma unroll
  for (int k = 0; k < 8; ++k) wHh[k] = make_frag_b(Whh, Hsz, nl, k * 32, lane);

  const float biasv = bih[nl] + bhh[nl];
  const float wfcv  = Wfc[nl];
  const float bfc0  = bfc[0];

  // Staging lane mapping: row = wave, 16B (8 f16) per lane.
  const int sr = tid >> 5;
  const int c8 = lane * 8;
  const _Float16* grow = h1 + (size_t)(bBase + sr) * Tlen * Hsz + c8;

  // Prime t = 0 into sIn[0] (async copy, pure f16 bytes).
  {
    uint64_t ga = (uint64_t)(uintptr_t)grow;
    uint32_t la = (uint32_t)(uintptr_t)&sIn[0][sr * HPAD + c8];
    asm volatile("global_load_async_to_lds_b128 %0, %1, off"
                 :: "v"(la), "v"(ga) : "memory");
  }

  int p = 0, hp = 0;
  for (int t = 0; t < Tlen; ++t) {
    // Issue async load of h1_{t+1} into the other buffer, then require the
    // t-step transfer to be complete (<=1 outstanding, in-order completion).
    if (t + 1 < Tlen) {
      uint64_t ga = (uint64_t)(uintptr_t)(grow + (size_t)(t + 1) * Hsz);
      uint32_t la = (uint32_t)(uintptr_t)&sIn[p ^ 1][sr * HPAD + c8];
      asm volatile("global_load_async_to_lds_b128 %0, %1, off"
                   :: "v"(la), "v"(ga) : "memory");
      asm volatile("s_wait_asynccnt 0x1" ::: "memory");
    } else {
      asm volatile("s_wait_asynccnt 0x0" ::: "memory");
    }
    __syncthreads();

    // Critical chain (recurrence): batch-preload all 8 A-fragments, pinned
    // with a sched_barrier so the 16 ds_load_b128 stay batched and the 8
    // recurrence WMMAs never hit a dscnt-0 stall. Projection fragments load
    // just-in-time inside the loop; each has a recurrence WMMA of slack and
    // only the non-critical chain waits on them.
    v16h ar[8];
#pragma unroll
    for (int k = 0; k < 8; ++k)
      ar[k] = load_frag_a(sH[hp], HPAD, mrow, k * 32, lane);
    __builtin_amdgcn_sched_barrier(0);

    v8f c0, c1;
#pragma unroll
    for (int r = 0; r < 8; ++r) { c0[r] = biasv; c1[r] = 0.f; }

#pragma unroll
    for (int k = 0; k < 8; ++k) {
      v16h pk = load_frag_a(sIn[p], HPAD, mrow, k * 32, lane);
      c1 = WMMA_F16(ar[k], wHh[k], c1);   // recurrence chain (never waits)
      c0 = WMMA_F16(pk, wIh[k], c0);      // projection chain
    }

    float vout[8];
#pragma unroll
    for (int r = 0; r < 8; ++r) {
      float v = fmaxf(c0[r] + c1[r], 0.f);                 // ReLU
      vout[r] = v;
      // FC head: butterfly-reduce v*wfc over the 16-lane half, then
      // cross-wave accumulate with LDS float atomics (ds_add_f32).
      float sv = v * wfcv;
      sv += __shfl_xor(sv, 1, 32);
      sv += __shfl_xor(sv, 2, 32);
      sv += __shfl_xor(sv, 4, 32);
      sv += __shfl_xor(sv, 8, 32);
      if ((lane & 15) == 0) atomicAdd(&sAcc[mbase + r], sv);
    }
    __syncthreads();   // all atomics done; old sH fully consumed

    const int hq = hp ^ 1;
#pragma unroll
    for (int r = 0; r < 8; ++r)
      sH[hq][(mbase + r) * HPAD + nl] = (_Float16)vout[r];

    if (wave == 0 && lane < 16) {
      out[(size_t)(bBase + lane) * Tlen + t] = sAcc[lane] + bfc0;
      sAcc[lane] = 0.f;
    }
    hp = hq;
    p ^= 1;
  }
}

// ---------------------------------------------------------------------------
// Inputs (setup_inputs order): x, W_ih0, W_hh0, b_ih0, b_hh0,
//                              W_ih1, W_hh1, b_ih1, b_hh1, W_fc, b_fc
// ---------------------------------------------------------------------------
extern "C" void kernel_launch(void* const* d_in, const int* in_sizes, int n_in,
                              void* d_out, int out_size, void* d_ws, size_t ws_size,
                              hipStream_t stream) {
  const float* x    = (const float*)d_in[0];
  const float* Wih0 = (const float*)d_in[1];
  const float* Whh0 = (const float*)d_in[2];
  const float* bih0 = (const float*)d_in[3];
  const float* bhh0 = (const float*)d_in[4];
  const float* Wih1 = (const float*)d_in[5];
  const float* Whh1 = (const float*)d_in[6];
  const float* bih1 = (const float*)d_in[7];
  const float* bhh1 = (const float*)d_in[8];
  const float* Wfc  = (const float*)d_in[9];
  const float* bfc  = (const float*)d_in[10];
  float* out = (float*)d_out;

  _Float16* h1 = (_Float16*)d_ws;   // B*T*H f16 = 64 MiB scratch

  dim3 grid(Bsz / 16), block(NTHREADS);
  rnn_layer0_kernel<<<grid, block, 0, stream>>>(x, Wih0, Whh0, bih0, bhh0, h1);
  rnn_layer1_kernel<<<grid, block, 0, stream>>>(h1, Wih1, Whh1, bih1, bhh1,
                                                Wfc, bfc, out);
}
